// TransformerEncoderLayer_40596030882100
// MI455X (gfx1250) — matrix operand order repaired and checked by reading, not compiled, not run on hardware
//
#include <hip/hip_runtime.h>

#define EMBED 1024
#define HEADS 16
#define HDIM  64
#define SEQ   2048
#define BATCH 4
#define FFDIM 4096
#define MTOK  (BATCH * SEQ)   // 8192 rows of tokens

typedef __attribute__((ext_vector_type(16))) __bf16 v16bf;
typedef __attribute__((ext_vector_type(8)))  __bf16 v8bf;
typedef __attribute__((ext_vector_type(8)))  float  v8f;

union Frag16 { v16bf v; v8bf h[2]; };

__device__ __forceinline__ v8f zero_v8f() {
  v8f z;
#pragma unroll
  for (int r = 0; r < 8; ++r) z[r] = 0.0f;
  return z;
}

__device__ __forceinline__ v8f wmma_bf16(v16bf a, v16bf b, v8f c) {
  // D = A(16x32 bf16) * B(32x16 bf16) + C(16x16 f32)
  return __builtin_amdgcn_wmma_f32_16x16x32_bf16(false, a, false, b, (short)0, c,
                                                 false, false);
}

// A-fragment (16xK row-major source): lane l holds row (l&15).
// Per 16-K block: lanes 0-15 take K+0..7, lanes 16-31 take K+8..15.
__device__ __forceinline__ v16bf load_afrag(const __bf16* rowptr, int k0, int half) {
  Frag16 f;
  f.h[0] = *reinterpret_cast<const v8bf*>(rowptr + k0 + 8 * half);
  f.h[1] = *reinterpret_cast<const v8bf*>(rowptr + k0 + 16 + 8 * half);
  return f.v;
}

// B-fragment from BT (NxK row-major): lane n holds column n of B.
// Per 16-K block: lanes 0-15 take K+0..7, lanes 16-31 take K+8..15.
__device__ __forceinline__ v16bf load_bfrag(const __bf16* rowptr, int k0, int half) {
  Frag16 f;
  f.h[0] = *reinterpret_cast<const v8bf*>(rowptr + k0 + 8 * half);
  f.h[1] = *reinterpret_cast<const v8bf*>(rowptr + k0 + 16 + 8 * half);
  return f.v;
}

// ---------------------------------------------------------------- GEMM
// C[M,N] = A[M,K](bf16) * BT[N,K](bf16)^T + bias, optional ReLU.
// Wave tile: 32x64 (2 A-frags x 4 B-frags -> 8 WMMA per K=32 step).
__global__ void __launch_bounds__(128)
gemm_bt_kernel(const __bf16* __restrict__ A, const __bf16* __restrict__ BT,
               const float* __restrict__ bias,
               float* __restrict__ Cf, __bf16* __restrict__ Cb,
               int M, int N, int K, int relu)
{
  const int lane = threadIdx.x & 31;
  const int wv   = threadIdx.x >> 5;
  const int half = lane >> 4;
  const int l16  = lane & 15;
  const int tiles_n = N >> 6;
  const int wave_id = blockIdx.x * 4 + wv;
  const int tm = wave_id / tiles_n;
  const int tn = wave_id - tm * tiles_n;
  const int m0 = tm << 5;
  const int n0 = tn << 6;

  const __bf16* a0p = A + (size_t)(m0 + l16) * K;
  const __bf16* a1p = A + (size_t)(m0 + 16 + l16) * K;
  const __bf16* bp[4];
#pragma unroll
  for (int j = 0; j < 4; ++j) bp[j] = BT + (size_t)(n0 + j * 16 + l16) * K;

  v8f c[2][4];
#pragma unroll
  for (int i = 0; i < 2; ++i)
#pragma unroll
    for (int j = 0; j < 4; ++j) c[i][j] = zero_v8f();

  for (int k0 = 0; k0 < K; k0 += 32) {
    __builtin_prefetch(a0p + k0 + 64, 0, 3);
    __builtin_prefetch(a1p + k0 + 64, 0, 3);
    v16bf a0 = load_afrag(a0p, k0, half);
    v16bf a1 = load_afrag(a1p, k0, half);
#pragma unroll
    for (int j = 0; j < 4; ++j) {
      v16bf b = load_bfrag(bp[j], k0, half);
      c[0][j] = wmma_bf16(a0, b, c[0][j]);
      c[1][j] = wmma_bf16(a1, b, c[1][j]);
    }
  }

#pragma unroll
  for (int j = 0; j < 4; ++j) {
    const int col = n0 + j * 16 + l16;
    const float bj = bias ? bias[col] : 0.0f;
#pragma unroll
    for (int i = 0; i < 2; ++i) {
#pragma unroll
      for (int r = 0; r < 8; ++r) {
        float v = c[i][j][r] + bj;
        if (relu) v = fmaxf(v, 0.0f);
        const size_t row = (size_t)(m0 + i * 16 + r + 8 * half);
        if (Cf) Cf[row * (size_t)N + col] = v;
        if (Cb) Cb[row * (size_t)N + col] = (__bf16)v;
      }
    }
  }
}

// ---------------------------------------------------------------- Attention
// One wave handles one 16-query tile of one (b,h); flash-style online softmax
// over 32-token K/V tiles. Qb/Kb: [B*S, E] bf16. Vt: [B,H,Dh,S] bf16.
__global__ void __launch_bounds__(128)
attn_kernel(const __bf16* __restrict__ Qb, const __bf16* __restrict__ Kb,
            const __bf16* __restrict__ Vt, __bf16* __restrict__ Cb)
{
  __shared__ __bf16 psh[4 * 16 * 32];
  const int lane = threadIdx.x & 31;
  const int wv   = threadIdx.x >> 5;
  const int half = lane >> 4;
  const int l16  = lane & 15;
  const int wave_id = blockIdx.x * 4 + wv;
  const int qt = wave_id & (SEQ / 16 - 1);
  const int bh = wave_id >> 7;
  const int h  = bh & (HEADS - 1);
  const int b  = bh >> 4;
  const int s0 = qt << 4;

  const __bf16* qrow = Qb + (size_t)(b * SEQ + s0 + l16) * EMBED + h * HDIM;
  const v16bf aq0 = load_afrag(qrow, 0, half);
  const v16bf aq1 = load_afrag(qrow, 32, half);

  float mst[8], lst[8];
  v8f ctx[4];
#pragma unroll
  for (int r = 0; r < 8; ++r) { mst[r] = -1e30f; lst[r] = 0.0f; }
#pragma unroll
  for (int d = 0; d < 4; ++d) ctx[d] = zero_v8f();

  __bf16* pw = psh + wv * (16 * 32);

  for (int kv0 = 0; kv0 < SEQ; kv0 += 32) {
    float sc[2][8];
#pragma unroll
    for (int j = 0; j < 2; ++j) {
      const __bf16* krow =
          Kb + (size_t)(b * SEQ + kv0 + j * 16 + l16) * EMBED + h * HDIM;
      v16bf bk0 = load_bfrag(krow, 0, half);
      v16bf bk1 = load_bfrag(krow, 32, half);
      v8f acc = zero_v8f();
      acc = wmma_bf16(aq0, bk0, acc);
      acc = wmma_bf16(aq1, bk1, acc);
#pragma unroll
      for (int r = 0; r < 8; ++r) sc[j][r] = acc[r] * 0.125f;  // 1/sqrt(64)
    }

    // per-row max across the 16 lanes of each half-wave
    float tmx[8];
#pragma unroll
    for (int r = 0; r < 8; ++r) tmx[r] = fmaxf(sc[0][r], sc[1][r]);
#pragma unroll
    for (int off = 1; off < 16; off <<= 1)
#pragma unroll
      for (int r = 0; r < 8; ++r)
        tmx[r] = fmaxf(tmx[r], __shfl_xor(tmx[r], off, 32));

    float alpha[8], rs[8];
#pragma unroll
    for (int r = 0; r < 8; ++r) {
      const float mn = fmaxf(mst[r], tmx[r]);
      alpha[r] = __expf(mst[r] - mn);
      mst[r] = mn;
      const float p0 = __expf(sc[0][r] - mn);
      const float p1 = __expf(sc[1][r] - mn);
      sc[0][r] = p0;
      sc[1][r] = p1;
      rs[r] = p0 + p1;
    }
#pragma unroll
    for (int off = 1; off < 16; off <<= 1)
#pragma unroll
      for (int r = 0; r < 8; ++r) rs[r] += __shfl_xor(rs[r], off, 32);
#pragma unroll
    for (int r = 0; r < 8; ++r) lst[r] = lst[r] * alpha[r] + rs[r];
#pragma unroll
    for (int d = 0; d < 4; ++d)
#pragma unroll
      for (int r = 0; r < 8; ++r) ctx[d][r] *= alpha[r];

    // spill P (16x32, C-layout) to LDS, reload in A-fragment layout
#pragma unroll
    for (int j = 0; j < 2; ++j)
#pragma unroll
      for (int r = 0; r < 8; ++r)
        pw[(r + 8 * half) * 32 + j * 16 + l16] = (__bf16)sc[j][r];
    asm volatile("s_wait_dscnt 0" ::: "memory");
    const v16bf pa = load_afrag(pw + l16 * 32, 0, half);

#pragma unroll
    for (int d = 0; d < 4; ++d) {
      const __bf16* vrow =
          Vt + (size_t)((b * HEADS + h) * HDIM + d * 16 + l16) * SEQ + kv0;
      v16bf bv = load_bfrag(vrow, 0, half);
      ctx[d] = wmma_bf16(pa, bv, ctx[d]);
    }
  }

#pragma unroll
  for (int d = 0; d < 4; ++d)
#pragma unroll
    for (int r = 0; r < 8; ++r) {
      const float v = ctx[d][r] / lst[r];
      const size_t row = (size_t)(b * SEQ + s0 + r + 8 * half);
      Cb[row * EMBED + h * HDIM + d * 16 + l16] = (__bf16)v;
    }
}

// ---------------------------------------------------------------- LayerNorm
// out = LN(X + R) * g + b ; writes f32 and/or bf16. One block per row.
__global__ void __launch_bounds__(256)
ln_residual_kernel(const float* __restrict__ X, const float* __restrict__ R,
                   const float* __restrict__ gam, const float* __restrict__ bet,
                   float* __restrict__ outF, __bf16* __restrict__ outB)
{
  __shared__ float red[16];
  const int row = blockIdx.x;
  const int t = threadIdx.x;
  const int lane = t & 31, wv = t >> 5;
  const float* xr = X + (size_t)row * EMBED;
  const float* rr = R + (size_t)row * EMBED;
  float v[4];
  float sum = 0.0f, sq = 0.0f;
#pragma unroll
  for (int i = 0; i < 4; ++i) {
    const int c = t + i * 256;
    v[i] = xr[c] + rr[c];
    sum += v[i];
    sq += v[i] * v[i];
  }
#pragma unroll
  for (int off = 1; off < 32; off <<= 1) {
    sum += __shfl_xor(sum, off, 32);
    sq  += __shfl_xor(sq,  off, 32);
  }
  if (lane == 0) { red[wv] = sum; red[8 + wv] = sq; }
  __syncthreads();
  float ts = 0.0f, tq = 0.0f;
#pragma unroll
  for (int w2 = 0; w2 < 8; ++w2) { ts += red[w2]; tq += red[8 + w2]; }
  const float mu  = ts * (1.0f / EMBED);
  const float var = tq * (1.0f / EMBED) - mu * mu;
  const float inv = rsqrtf(var + 1e-5f);
#pragma unroll
  for (int i = 0; i < 4; ++i) {
    const int c = t + i * 256;
    const float o = (v[i] - mu) * inv * gam[c] + bet[c];
    if (outF) outF[(size_t)row * EMBED + c] = o;
    if (outB) outB[(size_t)row * EMBED + c] = (__bf16)o;
  }
}

// ---------------------------------------------------------------- Converts
__global__ void cvt_f32_bf16_kernel(const float* __restrict__ in,
                                    __bf16* __restrict__ out, int n) {
  for (int i = blockIdx.x * blockDim.x + threadIdx.x; i < n;
       i += gridDim.x * blockDim.x)
    out[i] = (__bf16)in[i];
}

// W[K,N] f32 -> WT[N,K] bf16
__global__ void transpose_cvt_kernel(const float* __restrict__ W,
                                     __bf16* __restrict__ WT, int K, int N) {
  const int total = K * N;
  for (int i = blockIdx.x * blockDim.x + threadIdx.x; i < total;
       i += gridDim.x * blockDim.x) {
    const int n = i / K, k = i - n * K;
    WT[i] = (__bf16)W[(size_t)k * N + n];
  }
}

// Vb[B*S, E] bf16 -> Vt[B,H,Dh,S] bf16
__global__ void transpose_v_kernel(const __bf16* __restrict__ Vb,
                                   __bf16* __restrict__ Vt) {
  const int total = MTOK * EMBED;
  for (int i = blockIdx.x * blockDim.x + threadIdx.x; i < total;
       i += gridDim.x * blockDim.x) {
    const int row = i / EMBED;        // b*SEQ + s
    const int col = i - row * EMBED;  // h*HDIM + d
    const int b = row / SEQ, s2 = row - b * SEQ;
    const int h = col / HDIM, d = col - h * HDIM;
    Vt[(size_t)((b * HEADS + h) * HDIM + d) * SEQ + s2] = Vb[i];
  }
}

// ---------------------------------------------------------------- Launcher
extern "C" void kernel_launch(void* const* d_in, const int* in_sizes, int n_in,
                              void* d_out, int out_size, void* d_ws,
                              size_t ws_size, hipStream_t stream)
{
  (void)in_sizes; (void)n_in; (void)out_size; (void)ws_size;
  const float* x   = (const float*)d_in[0];
  const float* Wq  = (const float*)d_in[1];
  const float* bq  = (const float*)d_in[2];
  const float* Wk  = (const float*)d_in[3];
  const float* bk  = (const float*)d_in[4];
  const float* Wv  = (const float*)d_in[5];
  const float* bv  = (const float*)d_in[6];
  const float* Wo  = (const float*)d_in[7];
  const float* bo  = (const float*)d_in[8];
  const float* W1  = (const float*)d_in[9];
  const float* b1  = (const float*)d_in[10];
  const float* W2  = (const float*)d_in[11];
  const float* b2  = (const float*)d_in[12];
  const float* g1  = (const float*)d_in[13];
  const float* be1 = (const float*)d_in[14];
  const float* g2  = (const float*)d_in[15];
  const float* be2 = (const float*)d_in[16];
  float* out = (float*)d_out;

  char* w = (char*)d_ws;
  size_t off = 0;
  auto take = [&](size_t bytes) -> char* {
    char* p = w + off;
    off += (bytes + 255) & ~(size_t)255;
    return p;
  };

  __bf16* xb  = (__bf16*)take((size_t)MTOK * EMBED * 2);
  __bf16* WqT = (__bf16*)take((size_t)EMBED * EMBED * 2);
  __bf16* WkT = (__bf16*)take((size_t)EMBED * EMBED * 2);
  __bf16* WvT = (__bf16*)take((size_t)EMBED * EMBED * 2);
  __bf16* WoT = (__bf16*)take((size_t)EMBED * EMBED * 2);
  __bf16* W1T = (__bf16*)take((size_t)EMBED * FFDIM * 2);
  __bf16* W2T = (__bf16*)take((size_t)FFDIM * EMBED * 2);
  __bf16* Qb  = (__bf16*)take((size_t)MTOK * EMBED * 2);
  __bf16* Kb  = (__bf16*)take((size_t)MTOK * EMBED * 2);
  __bf16* Vb  = (__bf16*)take((size_t)MTOK * EMBED * 2);
  __bf16* Vt  = (__bf16*)take((size_t)MTOK * EMBED * 2);
  __bf16* Cb  = (__bf16*)take((size_t)MTOK * EMBED * 2);
  float*  Y1  = (float*)take((size_t)MTOK * EMBED * 4);
  float*  H1f = (float*)take((size_t)MTOK * EMBED * 4);
  __bf16* H1b = (__bf16*)take((size_t)MTOK * EMBED * 2);
  __bf16* FFb = (__bf16*)take((size_t)MTOK * FFDIM * 2);

  const dim3 blk128(128), blk256(256);
  const int cvt_grid = 2048;

  // 1) casts / transposes
  cvt_f32_bf16_kernel<<<cvt_grid, blk256, 0, stream>>>(x, xb, MTOK * EMBED);
  transpose_cvt_kernel<<<cvt_grid, blk256, 0, stream>>>(Wq, WqT, EMBED, EMBED);
  transpose_cvt_kernel<<<cvt_grid, blk256, 0, stream>>>(Wk, WkT, EMBED, EMBED);
  transpose_cvt_kernel<<<cvt_grid, blk256, 0, stream>>>(Wv, WvT, EMBED, EMBED);
  transpose_cvt_kernel<<<cvt_grid, blk256, 0, stream>>>(Wo, WoT, EMBED, EMBED);
  transpose_cvt_kernel<<<cvt_grid, blk256, 0, stream>>>(W1, W1T, EMBED, FFDIM);
  transpose_cvt_kernel<<<cvt_grid, blk256, 0, stream>>>(W2, W2T, FFDIM, EMBED);

  // 2) Q/K/V projections (bf16 out)
  const int grP = (MTOK / 32) * (EMBED / 64) / 4;  // 1024 blocks
  gemm_bt_kernel<<<grP, blk128, 0, stream>>>(xb, WqT, bq, nullptr, Qb,
                                             MTOK, EMBED, EMBED, 0);
  gemm_bt_kernel<<<grP, blk128, 0, stream>>>(xb, WkT, bk, nullptr, Kb,
                                             MTOK, EMBED, EMBED, 0);
  gemm_bt_kernel<<<grP, blk128, 0, stream>>>(xb, WvT, bv, nullptr, Vb,
                                             MTOK, EMBED, EMBED, 0);
  transpose_v_kernel<<<cvt_grid, blk256, 0, stream>>>(Vb, Vt);

  // 3) attention -> ctx (bf16)
  const int grA = BATCH * HEADS * (SEQ / 16) / 4;  // 2048 blocks
  attn_kernel<<<grA, blk128, 0, stream>>>(Qb, Kb, Vt, Cb);

  // 4) output projection (f32 out)
  gemm_bt_kernel<<<grP, blk128, 0, stream>>>(Cb, WoT, bo, Y1, nullptr,
                                             MTOK, EMBED, EMBED, 0);

  // 5) LN1(x + attn_out) -> f32 + bf16
  ln_residual_kernel<<<MTOK, blk256, 0, stream>>>(Y1, x, g1, be1, H1f, H1b);

  // 6) FF up + ReLU (bf16 out)
  const int grF1 = (MTOK / 32) * (FFDIM / 64) / 4;  // 4096 blocks
  gemm_bt_kernel<<<grF1, blk128, 0, stream>>>(H1b, W1T, b1, nullptr, FFb,
                                              MTOK, FFDIM, EMBED, 1);

  // 7) FF down (f32 out)
  gemm_bt_kernel<<<grP, blk128, 0, stream>>>(FFb, W2T, b2, Y1, nullptr,
                                             MTOK, EMBED, FFDIM, 0);

  // 8) LN2(h1 + ff) -> final f32 output
  ln_residual_kernel<<<MTOK, blk256, 0, stream>>>(Y1, H1f, g2, be2, out, nullptr);
}
